// MultiHeadedAttention_90529320665928
// MI455X (gfx1250) — matrix operand order repaired and checked by reading, not compiled, not run on hardware
//
#include <hip/hip_runtime.h>
#include <hip/hip_bf16.h>

// ---------------------------------------------------------------------------
// MHA forward on gfx1250 (wave32, WMMA bf16 16x16x32, f32 accumulate)
//   B=4, S=2048, D=1024, H=16, HD=64
// Async global->LDS staging (ASYNCcnt) + double-buffered LDS in the GEMMs.
// ---------------------------------------------------------------------------

typedef __bf16 bf16x16 __attribute__((ext_vector_type(16)));
typedef float  f32x8   __attribute__((ext_vector_type(8)));

union AFrag { bf16x16 v; uint4 q[2]; };   // 16 bf16 = 32 bytes = 8 VGPRs

#define NB 4
#define NS 2048
#define ND 1024
#define NH 16
#define NHD 64

__device__ __forceinline__ __bf16 f2b(float f) {
    union { float f; unsigned u; } a; a.f = f;
    unsigned r = (a.u + 0x7FFFu + ((a.u >> 16) & 1u)) >> 16;   // RNE
    union { unsigned short s; __bf16 b; } o; o.s = (unsigned short)r;
    return o.b;
}

// LDS generic pointers keep the workgroup-relative byte offset in the low
// 32 bits (ISA 10.2: LDS aperture address = {aperture_hi, offset}).
__device__ __forceinline__ unsigned lds_off(const void* p) {
    return (unsigned)(unsigned long long)p;
}

// Async DMA: 16B per lane, memory -> LDS, tracked by ASYNCcnt (no VGPR
// round-trip, no loadcnt stall; overlaps with WMMA issue).
__device__ __forceinline__ void async_copy_b128(unsigned lds, const void* g) {
    asm volatile("global_load_async_to_lds_b128 %0, %1, off"
                 :: "v"(lds), "v"((unsigned long long)g) : "memory");
}
__device__ __forceinline__ void wait_async0() {
    asm volatile("s_wait_asynccnt 0" ::: "memory");
}

// ---------------------------------------------------------------------------
// Elementwise f32 -> bf16 (x matrix), vectorized 4-wide.
// ---------------------------------------------------------------------------
__global__ __launch_bounds__(256) void cvt_x_kernel(const float* __restrict__ X,
                                                    __bf16* __restrict__ Y, int n4) {
    int i = blockIdx.x * 256 + threadIdx.x;
    if (i >= n4) return;
    float4 v = ((const float4*)X)[i];
    union { __bf16 b[4]; uint2 u; } o;
    o.b[0] = f2b(v.x); o.b[1] = f2b(v.y); o.b[2] = f2b(v.z); o.b[3] = f2b(v.w);
    ((uint2*)Y)[i] = o.u;
}

// ---------------------------------------------------------------------------
// Transpose + convert weights: Wt[n][k] = bf16(W[k][n]).  One-time cost so the
// GEMM B-fragment (lane=col, contiguous k-chunk) is a straight 16B LDS read.
// ---------------------------------------------------------------------------
__global__ __launch_bounds__(256) void cvt_wT_kernel(const float* __restrict__ W,
                                                     __bf16* __restrict__ Wt,
                                                     int K, int N) {
    int idx = blockIdx.x * 256 + threadIdx.x;
    if (idx >= K * N) return;
    int k = idx % K, n = idx / K;          // write side (Wt) coalesced
    Wt[(size_t)n * K + k] = f2b(W[(size_t)k * N + n]);
}

// ---------------------------------------------------------------------------
// Tiled GEMM  C = A[M,K] x Bt[N,K]^T + bias.
//   mode 0: QKV epilogue -> scatter bf16 into Q/K/V [B,H,S,HD]
//   mode 1: f32 epilogue -> Cout[M,N]
// Block 256 threads = 8 waves in 4x2; tile 128x128x32; wave tile 32x64.
// Double-buffered LDS; next K-tile DMA'd asynchronously during compute.
// ---------------------------------------------------------------------------
#define BM 128
#define BN 128
#define BK 32
#define LDT 40   // padded LDS row (elements); 80B row stride -> 16B aligned

__global__ __launch_bounds__(256) void gemm_bf16_kernel(
    const __bf16* __restrict__ A, const __bf16* __restrict__ Bt,
    const float* __restrict__ bias, int M, int N, int K, int mode,
    __bf16* __restrict__ Qh, __bf16* __restrict__ Kh, __bf16* __restrict__ Vh,
    float* __restrict__ Cout)
{
    __shared__ __align__(16) __bf16 As[2][BM][LDT];
    __shared__ __align__(16) __bf16 Bs[2][BN][LDT];

    const int tid  = threadIdx.x;
    const int lane = tid & 31, wid = tid >> 5;
    const int wr = wid & 3, wc = wid >> 2;        // 4 row-groups x 2 col-groups
    const int ln = lane & 15, half = lane >> 4;
    const int m0 = blockIdx.y * BM, n0 = blockIdx.x * BN;

    f32x8 acc[2][4] = {};

    const int srow = tid >> 1, schunk = (tid & 1) * 16;   // 16 elems x2 mats / thread
    const __bf16* agBase = A  + (size_t)(m0 + srow) * K + schunk;
    const __bf16* bgBase = Bt + (size_t)(n0 + srow) * K + schunk;

    // prologue: DMA tile 0 into buffer 0
    async_copy_b128(lds_off(&As[0][srow][schunk]),     agBase);
    async_copy_b128(lds_off(&As[0][srow][schunk + 8]), agBase + 8);
    async_copy_b128(lds_off(&Bs[0][srow][schunk]),     bgBase);
    async_copy_b128(lds_off(&Bs[0][srow][schunk + 8]), bgBase + 8);
    wait_async0();
    __syncthreads();

    int buf = 0;
    for (int k0 = 0; k0 < K; k0 += BK) {
        if (k0 + BK < K) {                         // DMA next tile into other buffer
            const __bf16* ag = agBase + k0 + BK;
            const __bf16* bg = bgBase + k0 + BK;
            async_copy_b128(lds_off(&As[buf ^ 1][srow][schunk]),     ag);
            async_copy_b128(lds_off(&As[buf ^ 1][srow][schunk + 8]), ag + 8);
            async_copy_b128(lds_off(&Bs[buf ^ 1][srow][schunk]),     bg);
            async_copy_b128(lds_off(&Bs[buf ^ 1][srow][schunk + 8]), bg + 8);
        }

        AFrag fa[2], fb[4];
        for (int i = 0; i < 2; i++) {              // A: lane=row, k-chunks 8h / 16+8h
            int r = wr * 32 + i * 16 + ln;
            fa[i].q[0] = *(const uint4*)&As[buf][r][8 * half];
            fa[i].q[1] = *(const uint4*)&As[buf][r][16 + 8 * half];
        }
        for (int j = 0; j < 4; j++) {              // B: lane=col, k-chunks 8h / 16+8h
            int c = wc * 64 + j * 16 + ln;
            fb[j].q[0] = *(const uint4*)&Bs[buf][c][8 * half];
            fb[j].q[1] = *(const uint4*)&Bs[buf][c][16 + 8 * half];
        }
        for (int i = 0; i < 2; i++)
            for (int j = 0; j < 4; j++)
                acc[i][j] = __builtin_amdgcn_wmma_f32_16x16x32_bf16(
                    false, fa[i].v, false, fb[j].v, (short)0, acc[i][j], false, false);

        wait_async0();      // next tile landed in LDS (overlapped with WMMAs above)
        __syncthreads();
        buf ^= 1;
    }

    for (int i = 0; i < 2; i++)
        for (int j = 0; j < 4; j++) {
            int n = n0 + wc * 64 + j * 16 + ln;
            float bv = bias[n];
            for (int r = 0; r < 8; r++) {          // D: row = r + 8*half, col = ln
                int m = m0 + wr * 32 + i * 16 + r + 8 * half;
                float v = acc[i][j][r] + bv;
                if (mode == 1) {
                    Cout[(size_t)m * N + n] = v;
                } else {
                    int which = n >> 10, d = n & 1023;
                    int h = d >> 6, hd = d & 63;
                    int b = m >> 11, s = m & 2047;
                    size_t di = ((size_t)(b * NH + h) * NS + s) * NHD + hd;
                    __bf16 bb = f2b(v);
                    if (which == 0)      Qh[di] = bb;
                    else if (which == 1) Kh[di] = bb;
                    else                 Vh[di] = bb;
                }
            }
        }
}

// ---------------------------------------------------------------------------
// Flash-attention: grid (B*H, S/64); block 128 = 4 waves x 16 query rows.
// Keys streamed in 32-wide tiles (K via async DMA); online softmax; O in f32.
// Causal: block y processes key tiles 0 .. 2*(y+1)-1, per-element mask.
// ---------------------------------------------------------------------------
__global__ __launch_bounds__(128) void attn_kernel(
    const __bf16* __restrict__ Qh, const __bf16* __restrict__ Kh,
    const __bf16* __restrict__ Vh, __bf16* __restrict__ Yh)
{
    __shared__ __align__(16) __bf16 Ks [32][72];    // [key][hd], 144B rows
    __shared__ __align__(16) __bf16 Vst[64][40];    // [hd][key] (transposed)
    __shared__ __align__(16) __bf16 Ps [4][16][40]; // per-wave P relayout buffer

    const int tid  = threadIdx.x;
    const int lane = tid & 31, wid = tid >> 5;
    const int ln = lane & 15, half = lane >> 4;
    const int bh = blockIdx.x;                      // 0..63
    const int b = bh >> 4, h = bh & 15;
    const int qbase = blockIdx.y * 64 + wid * 16;

    // Q fragments (HD=64 -> two K=32 A-fragments), scale folded at score stage
    AFrag fq[2];
    {
        const __bf16* qp = Qh + ((size_t)bh * NS + qbase + ln) * NHD;
        for (int t = 0; t < 2; t++) {
            fq[t].q[0] = *(const uint4*)(qp + t * 32 + 8 * half);
            fq[t].q[1] = *(const uint4*)(qp + t * 32 + 16 + 8 * half);
        }
    }

    f32x8 o[4] = {};
    float mrun[8], lrun[8];
    for (int r = 0; r < 8; r++) { mrun[r] = -1e30f; lrun[r] = 0.f; }

    const int nkt = 2 * blockIdx.y + 2;             // uniform in block
    for (int kt = 0; kt < nkt; kt++) {
        const int kbase = kt * 32;
        {   // stage K rows via async DMA; V via load + transposed store
            int key = tid >> 2, chunk = (tid & 3) * 16;
            const __bf16* kg = Kh + ((size_t)bh * NS + kbase + key) * NHD + chunk;
            async_copy_b128(lds_off(&Ks[key][chunk]),     kg);
            async_copy_b128(lds_off(&Ks[key][chunk + 8]), kg + 8);
            const __bf16* vg = Vh + ((size_t)bh * NS + kbase + key) * NHD + chunk;
            union { uint4 u; __bf16 e[8]; } va, vb;
            va.u = *(const uint4*)vg;
            vb.u = *(const uint4*)(vg + 8);
            for (int e = 0; e < 8; e++) {
                Vst[chunk + e][key]     = va.e[e];
                Vst[chunk + 8 + e][key] = vb.e[e];
            }
        }
        wait_async0();
        __syncthreads();

        // scores S[16 x 32] = Q[16 x 64] . K^T : two 16x16 tiles, 2 WMMAs each
        f32x8 sc[2] = {};
        for (int t = 0; t < 2; t++)
            for (int kh = 0; kh < 2; kh++) {
                AFrag fk;   // B operand: lane=key col, hd chunks 8h / 16+8h
                fk.q[0] = *(const uint4*)&Ks[t * 16 + ln][kh * 32 + 8 * half];
                fk.q[1] = *(const uint4*)&Ks[t * 16 + ln][kh * 32 + 16 + 8 * half];
                sc[t] = __builtin_amdgcn_wmma_f32_16x16x32_bf16(
                    false, fq[kh].v, false, fk.v, (short)0, sc[t], false, false);
            }

        // scale 1/sqrt(64), causal mask, online softmax (rows live in lane-halves)
        for (int r = 0; r < 8; r++) {
            int qrow = qbase + r + 8 * half;
            float s0 = sc[0][r] * 0.125f;
            float s1 = sc[1][r] * 0.125f;
            if (kbase + ln > qrow)      s0 = -1e30f;
            if (kbase + 16 + ln > qrow) s1 = -1e30f;
            float mx = fmaxf(s0, s1);
            for (int d = 1; d < 16; d <<= 1) mx = fmaxf(mx, __shfl_xor(mx, d));
            float mnew  = fmaxf(mrun[r], mx);
            float alpha = __expf(mrun[r] - mnew);
            float p0 = __expf(s0 - mnew);
            float p1 = __expf(s1 - mnew);
            float rs = p0 + p1;
            for (int d = 1; d < 16; d <<= 1) rs += __shfl_xor(rs, d);
            lrun[r] = lrun[r] * alpha + rs;
            mrun[r] = mnew;
            for (int j = 0; j < 4; j++) o[j][r] *= alpha;
            // park P in per-wave LDS (D-layout -> A-layout relayout)
            Ps[wid][r + 8 * half][ln]      = f2b(p0);
            Ps[wid][r + 8 * half][16 + ln] = f2b(p1);
        }
        // per-wave DS ops are in-order: safe to reload P as an A-fragment
        AFrag fp;
        fp.q[0] = *(const uint4*)&Ps[wid][ln][8 * half];
        fp.q[1] = *(const uint4*)&Ps[wid][ln][16 + 8 * half];
        for (int j = 0; j < 4; j++) {               // O[16x64] += P[16x32] . V[32x64]
            AFrag fv;                                // B operand from transposed V
            fv.q[0] = *(const uint4*)&Vst[j * 16 + ln][8 * half];
            fv.q[1] = *(const uint4*)&Vst[j * 16 + ln][16 + 8 * half];
            o[j] = __builtin_amdgcn_wmma_f32_16x16x32_bf16(
                false, fp.v, false, fv.v, (short)0, o[j], false, false);
        }
        __syncthreads();
    }

    // normalize and write bf16 attention output in [B,S,D] for the proj GEMM
    for (int j = 0; j < 4; j++)
        for (int r = 0; r < 8; r++) {
            int srow = qbase + r + 8 * half;
            float v = o[j][r] / lrun[r];
            Yh[((size_t)b * NS + srow) * ND + h * NHD + j * 16 + ln] = f2b(v);
        }
}

// ---------------------------------------------------------------------------
// Launch: cvt -> QKV GEMM -> flash attention -> proj GEMM.  Workspace ~75 MB.
// ---------------------------------------------------------------------------
extern "C" void kernel_launch(void* const* d_in, const int* in_sizes, int n_in,
                              void* d_out, int out_size, void* d_ws, size_t ws_size,
                              hipStream_t stream) {
    const float* x  = (const float*)d_in[0];   // [4,2048,1024]
    const float* Wa = (const float*)d_in[1];   // [1024,3072]
    const float* ba = (const float*)d_in[2];   // [3072]
    const float* Wp = (const float*)d_in[3];   // [1024,1024]
    const float* bp = (const float*)d_in[4];   // [1024]
    float* out = (float*)d_out;                // [4,2048,1024] f32

    const size_t MSZ  = (size_t)NB * NS * ND;           // 8,388,608
    char* ws = (char*)d_ws;
    __bf16* xb  = (__bf16*)(ws);                        // 16 MB (reused as Y)
    __bf16* WaT = (__bf16*)(ws + MSZ * 2);              //  6 MB
    __bf16* WpT = (__bf16*)(ws + MSZ * 2 + (size_t)ND * 3 * ND * 2);
    __bf16* Qh  = (__bf16*)(ws + MSZ * 2 + (size_t)ND * 4 * ND * 2);
    __bf16* Kh  = Qh + MSZ;
    __bf16* Vh  = Kh + MSZ;
    __bf16* Yh  = xb;                                   // x dead after QKV GEMM

    // 1) convert x to bf16
    cvt_x_kernel<<<(int)(MSZ / 4 / 256), 256, 0, stream>>>(x, xb, (int)(MSZ / 4));
    // 2) transpose+convert weights
    cvt_wT_kernel<<<(ND * 3 * ND) / 256, 256, 0, stream>>>(Wa, WaT, ND, 3 * ND);
    cvt_wT_kernel<<<(ND * ND) / 256, 256, 0, stream>>>(Wp, WpT, ND, ND);
    // 3) QKV GEMM + head scatter
    gemm_bf16_kernel<<<dim3(3 * ND / BN, NB * NS / BM), 256, 0, stream>>>(
        xb, WaT, ba, NB * NS, 3 * ND, ND, 0, Qh, Kh, Vh, nullptr);
    // 4) causal flash attention
    attn_kernel<<<dim3(NB * NH, NS / 64), 128, 0, stream>>>(Qh, Kh, Vh, Yh);
    // 5) output projection (f32 out)
    gemm_bf16_kernel<<<dim3(ND / BN, NB * NS / BM), 256, 0, stream>>>(
        Yh, WpT, bp, NB * NS, ND, ND, 1, nullptr, nullptr, nullptr, out);
}
